// LanguageModel_29059748725215
// MI455X (gfx1250) — matrix operand order repaired and checked by reading, not compiled, not run on hardware
//
#include <hip/hip_runtime.h>
#include <hip/hip_bf16.h>
#include <math.h>

// Problem constants (from reference)
#define B_  4
#define S_  512
#define H_  768
#define L_  12
#define NH_ 12
#define DH_ 64
#define F_  3072
#define V_  21128
#define NT_ (B_*S_)
#define NEGBIG (-1.0e9f)
#define ATTN_SCALE 0.125f   // 1/sqrt(64)

typedef __attribute__((ext_vector_type(16))) __bf16 v16bf;
typedef __attribute__((ext_vector_type(8)))  float  v8f;
typedef unsigned int u32;
typedef unsigned long long u64;
typedef __attribute__((ext_vector_type(4))) unsigned int uint32x4;
typedef __attribute__((ext_vector_type(4))) int          int32x4;
typedef __attribute__((ext_vector_type(8))) int          int32x8;

// Use the Tensor Data Mover when the toolchain exposes it (device pass only).
#if defined(__HIP_DEVICE_COMPILE__)
#  if __has_builtin(__builtin_amdgcn_tensor_load_to_lds) && __has_builtin(__builtin_amdgcn_s_wait_tensorcnt)
#    define USE_TDM 1
#  endif
#endif
#ifndef USE_TDM
#  define USE_TDM 0
#endif

// ---------------------------------------------------------------- helpers
__device__ __forceinline__ u32 pack2bf(float f0, float f1) {
    unsigned short a = __builtin_bit_cast(unsigned short, (__bf16)f0);
    unsigned short b = __builtin_bit_cast(unsigned short, (__bf16)f1);
    return (u32)a | ((u32)b << 16);
}

__device__ __forceinline__ float block_reduce_sum(float v, float* red) {
    int t = threadIdx.x;
    red[t] = v; __syncthreads();
    for (int o = 128; o > 0; o >>= 1) { if (t < o) red[t] += red[t + o]; __syncthreads(); }
    float r = red[0]; __syncthreads();
    return r;
}

__device__ __forceinline__ float block_reduce_max(float v, float* red) {
    int t = threadIdx.x;
    red[t] = v; __syncthreads();
    for (int o = 128; o > 0; o >>= 1) { if (t < o) red[t] = fmaxf(red[t], red[t + o]); __syncthreads(); }
    float r = red[0]; __syncthreads();
    return r;
}

#if USE_TDM
// 2D TDM load: tile (tile_x contiguous elems) x (tile_y rows, row stride = stride
// fp32 elements) from gsrc into LDS at lds_byte (linear, row-major tile_x).
// tensor_x/tensor_y: remaining in-bounds extents (beyond -> zero-filled).
__device__ __forceinline__ void tdm_load_tile(const float* gsrc, u32 lds_byte,
                                              u32 tile_x, u32 tile_y,
                                              u64 stride, u32 tensor_x, u32 tensor_y) {
    u64 ga = (u64)(uintptr_t)gsrc;
    uint32x4 g0;
    g0.x = 1u;                                   // count=1, user descriptor
    g0.y = lds_byte;                             // lds_addr
    g0.z = (u32)(ga & 0xFFFFFFFFull);            // global_addr[31:0]
    g0.w = (u32)((ga >> 32) & 0x1FFFFFFull) | (2u << 30);  // global_addr[56:32] | type=2

    int32x8 g1;
    g1[0] = (int)(2u << 16);                     // workgroup_mask=0, data_size=2 (4B)
    g1[1] = (int)((tensor_x & 0xFFFFu) << 16);   // tensor_dim0[15:0]
    g1[2] = (int)((tensor_x >> 16) | ((tensor_y & 0xFFFFu) << 16));
    g1[3] = (int)((tensor_y >> 16) | (tile_x << 16));     // tile_dim0
    g1[4] = (int)(tile_y & 0xFFFFu);             // tile_dim1 (tile_dim2 = 0)
    g1[5] = (int)(u32)(stride & 0xFFFFFFFFull);  // tensor_dim0_stride[31:0]
    g1[6] = (int)(u32)((stride >> 32) & 0xFFFFull);
    g1[7] = 0;

    int32x4 z4 = {0, 0, 0, 0};
#if __clang_major__ >= 23
    int32x8 z8 = {0, 0, 0, 0, 0, 0, 0, 0};
    __builtin_amdgcn_tensor_load_to_lds(g0, g1, z4, z4, z8, 0);
#else
    __builtin_amdgcn_tensor_load_to_lds(g0, g1, z4, z4, 0);
#endif
}
#endif

// ---------------------------------------------------------------- GEMM (WMMA bf16, TDM-staged B)
// C[m,n] = act( A[m,k] * B[k,n] + bias[n] + R[m,n] )
// A: fp32, element (m,k) at A + m*lda + k            (k contiguous)
// B: fp32, element (k,n) at B + k*ldbk + n*ldbn ; requires ldbn==1 or ldbk==1
// Block: 128 threads = 4 waves; block tile 128(M) x 64(N); wave tile 32x64.
// Requires M % 128 == 0 and K % 32 == 0 (true for every call below).
struct GemmParams {
    const float* A;  long long lda,  aStrB, aStrH;
    const float* Bm; long long ldbk, ldbn, bStrB, bStrH;
    float*       C;  long long ldc,  cStrB, cStrH;
    const float* bias;               // per-n, may be null
    const float* R;  long long ldr;  // residual, may be null
    int M, N, K;
    int zdiv;                        // NH for attention batching, else 1
    int gelu;                        // 1 -> exact GELU on output
};

struct W8 { u32 w[8]; };             // one packed bf16 B fragment slice per lane

__global__ __launch_bounds__(128) void gemm_wmma_bf16(GemmParams p) {
#if USE_TDM
    __shared__ float bt[2][2048];    // double-buffered raw fp32 B tile (16 KB)
#endif
    __shared__ u32 btp[1024];        // bf16-packed, fragment-ready B tile (4 KB)
                                     // btp[((t*32+lane)<<3)+vi] = pair(k=16*(lane>>4)+2vi, n=t*16+(lane&15))

    const int tid  = threadIdx.x;
    const int lane = tid & 31;
    const int wave = tid >> 5;
    const int half = lane >> 4;
    const int lrow = lane & 15;

    const int z  = blockIdx.z;
    const int zb = z / p.zdiv;
    const int zh = z - zb * p.zdiv;

    const float* A  = p.A  + (long long)zb * p.aStrB + (long long)zh * p.aStrH;
    const float* Bm = p.Bm + (long long)zb * p.bStrB + (long long)zh * p.bStrH;
    float*       C  = p.C  + (long long)zb * p.cStrB + (long long)zh * p.cStrH;

    const int m0 = (blockIdx.x * 4 + wave) * 32;    // wave's 32-row stripe
    const int n0 = blockIdx.y * 64;
    const int btrans = (p.ldbk == 1) ? 1 : 0;       // 1: K contiguous in memory
    const long long bstride = btrans ? p.ldbn : p.ldbk;
    const u32 rem_n = (u32)((p.N - n0) > 64 ? 64 : (p.N - n0));

    v8f zero = {0.f,0.f,0.f,0.f,0.f,0.f,0.f,0.f};
    v8f acc[2][4] = {{zero, zero, zero, zero}, {zero, zero, zero, zero}};

    int cur = 0;
#if USE_TDM
    if (wave == 0) {
        const float* ts0 = btrans ? (Bm + (long long)n0 * p.ldbn)
                                  : (Bm + (long long)n0);
        if (btrans)
            tdm_load_tile(ts0, (u32)(uintptr_t)(&bt[0][0]), 32u, 64u,
                          (u64)bstride, (u32)p.K, rem_n);
        else
            tdm_load_tile(ts0, (u32)(uintptr_t)(&bt[0][0]), 64u, 32u,
                          (u64)bstride, rem_n, (u32)p.K);
    }
#endif

    for (int k0 = 0; k0 < p.K; k0 += 32) {
#if USE_TDM
        if (wave == 0) __builtin_amdgcn_s_wait_tensorcnt(0);
#endif
        __syncthreads();   // (a) bt[cur] DMA visible; (b) previous btp readers done
#if USE_TDM
        if (wave == 0 && (k0 + 32) < p.K) {
            int kn = k0 + 32;
            const float* tsn = btrans ? (Bm + (long long)kn + (long long)n0 * p.ldbn)
                                      : (Bm + (long long)kn * p.ldbk + (long long)n0);
            if (btrans)
                tdm_load_tile(tsn, (u32)(uintptr_t)(&bt[cur ^ 1][0]), 32u, 64u,
                              (u64)bstride, (u32)(p.K - kn), rem_n);
            else
                tdm_load_tile(tsn, (u32)(uintptr_t)(&bt[cur ^ 1][0]), 64u, 32u,
                              (u64)bstride, rem_n, (u32)(p.K - kn));
        }
        // pack pass: fp32 tile -> bf16 fragment-ready words (once per block)
        for (int id = tid; id < 1024; id += 128) {
            int vi = id & 7, ln = (id >> 3) & 31, tt = id >> 8;
            int kk = (ln >> 4) * 16 + 2 * vi;
            int cr = tt * 16 + (ln & 15);
            float f0, f1;
            if (btrans) { f0 = bt[cur][cr * 32 + kk];  f1 = bt[cur][cr * 32 + kk + 1]; }
            else        { f0 = bt[cur][kk * 64 + cr];  f1 = bt[cur][(kk + 1) * 64 + cr]; }
            btp[id] = pack2bf(f0, f1);
        }
#else
        // fallback: global -> bf16 fragment-ready words directly
        {
            const float* tileStart = btrans
                ? (Bm + (long long)k0 + (long long)n0 * p.ldbn)
                : (Bm + (long long)k0 * p.ldbk + (long long)n0);
            for (int id = tid; id < 1024; id += 128) {
                int vi = id & 7, ln = (id >> 3) & 31, tt = id >> 8;
                int kk = (ln >> 4) * 16 + 2 * vi;
                int cr = tt * 16 + (ln & 15);
                float f0 = 0.f, f1 = 0.f;
                if ((u32)cr < rem_n) {
                    if (btrans) {
                        const float* bp = tileStart + (long long)cr * bstride + kk;
                        f0 = bp[0]; f1 = bp[1];
                    } else {
                        const float* bp = tileStart + (long long)kk * bstride + cr;
                        f0 = bp[0]; f1 = bp[bstride];
                    }
                }
                btp[id] = pack2bf(f0, f1);
            }
        }
#endif
        __syncthreads();   // btp ready

        // ---------------- A fragments: 2 sub-tiles of 16x32 (contiguous k order: k = 16*half + s)
        v16bf fa[2];
        #pragma unroll
        for (int sub = 0; sub < 2; ++sub) {
            const float* arow = A + (long long)(m0 + sub * 16 + lrow) * p.lda + k0;
            __builtin_prefetch(arow + 32, 0, 1);   // next K tile -> global_prefetch_b8
            #pragma unroll
            for (int vi = 0; vi < 8; ++vi) {
                int kk = half * 16 + 2 * vi;
                fa[sub][2*vi]   = (__bf16)arow[kk];
                fa[sub][2*vi+1] = (__bf16)arow[kk + 1];
            }
        }

        // ---------------- 4 B fragments (2 x ds_load_b128 each), 8 WMMAs
        #pragma unroll
        for (int t = 0; t < 4; ++t) {
            W8 wv;
            #pragma unroll
            for (int vi = 0; vi < 8; ++vi)
                wv.w[vi] = btp[((t * 32 + lane) << 3) + vi];
            v16bf fb = __builtin_bit_cast(v16bf, wv);
            acc[0][t] = __builtin_amdgcn_wmma_f32_16x16x32_bf16(
                false, fa[0], false, fb, (short)0, acc[0][t], false, false);
            acc[1][t] = __builtin_amdgcn_wmma_f32_16x16x32_bf16(
                false, fa[1], false, fb, (short)0, acc[1][t], false, false);
        }
        cur ^= 1;
    }

    // ---------------- store C (16x16 C layout: row = vi + 8*half, col = lrow)
    #pragma unroll
    for (int sub = 0; sub < 2; ++sub) {
        #pragma unroll
        for (int t = 0; t < 4; ++t) {
            int col = n0 + t * 16 + lrow;
            if (col >= p.N) continue;
            float bv = p.bias ? p.bias[col] : 0.f;
            #pragma unroll
            for (int vi = 0; vi < 8; ++vi) {
                int row = m0 + sub * 16 + half * 8 + vi;
                float val = acc[sub][t][vi] + bv;
                if (p.R)    val += p.R[(long long)row * p.ldr + col];
                if (p.gelu) val = 0.5f * val * (1.f + erff(val * 0.70710678118654752f));
                C[(long long)row * p.ldc + col] = val;
            }
        }
    }
}

// ---------------------------------------------------------------- embedding + LN
__global__ __launch_bounds__(256) void embed_ln_kernel(
    const int* __restrict__ x, const float* __restrict__ we,
    const float* __restrict__ pe, const float* __restrict__ te,
    const float* __restrict__ g, const float* __restrict__ b,
    float* __restrict__ h) {
    __shared__ float red[256];
    const int tok = blockIdx.x;                 // 0..NT-1
    const int s   = tok % S_;
    const long long id = x[tok];
    float local[3];
    float sum = 0.f;
    int i = 0;
    for (int c = threadIdx.x; c < H_; c += 256, ++i) {
        float e = we[id * H_ + c] + pe[(long long)s * H_ + c] + te[c];
        local[i] = e; sum += e;
    }
    float mean = block_reduce_sum(sum, red) * (1.f / H_);
    float vs = 0.f;
    i = 0;
    for (int c = threadIdx.x; c < H_; c += 256, ++i) {
        float d = local[i] - mean; vs += d * d;
    }
    float var  = block_reduce_sum(vs, red) * (1.f / H_);
    float rstd = rsqrtf(var + 1e-12f);
    i = 0;
    for (int c = threadIdx.x; c < H_; c += 256, ++i)
        h[(long long)tok * H_ + c] = (local[i] - mean) * rstd * g[c] + b[c];
}

// ---------------------------------------------------------------- LayerNorm
__global__ __launch_bounds__(256) void ln_kernel(
    const float* __restrict__ in, const float* __restrict__ g,
    const float* __restrict__ b, float* __restrict__ out) {
    __shared__ float red[256];
    const int tok = blockIdx.x;
    float local[3];
    float sum = 0.f;
    int i = 0;
    for (int c = threadIdx.x; c < H_; c += 256, ++i) {
        float e = in[(long long)tok * H_ + c];
        local[i] = e; sum += e;
    }
    float mean = block_reduce_sum(sum, red) * (1.f / H_);
    float vs = 0.f;
    i = 0;
    for (int c = threadIdx.x; c < H_; c += 256, ++i) {
        float d = local[i] - mean; vs += d * d;
    }
    float var  = block_reduce_sum(vs, red) * (1.f / H_);
    float rstd = rsqrtf(var + 1e-12f);
    i = 0;
    for (int c = threadIdx.x; c < H_; c += 256, ++i)
        out[(long long)tok * H_ + c] = (local[i] - mean) * rstd * g[c] + b[c];
}

// ---------------------------------------------------------------- masked softmax (in place)
__global__ __launch_bounds__(256) void softmax_mask_kernel(
    float* __restrict__ scores, const int* __restrict__ tl) {
    __shared__ float red[256];
    const int row = blockIdx.x;                 // b*NH*S + hd*S + i
    const int i   = row % S_;
    const int b   = row / (NH_ * S_);
    const int t   = tl[b];
    float* sr = scores + (long long)row * S_;

    float vals[2];
    float vmax = -3.4e38f;
    int c = 0;
    for (int j = threadIdx.x; j < S_; j += 256, ++c) {
        bool mask = (i < t) ? (j < t) : ((j < t) || (j >= i));
        float v = sr[j] * ATTN_SCALE + (mask ? 0.f : NEGBIG);
        vals[c] = v;
        vmax = fmaxf(vmax, v);
    }
    vmax = block_reduce_max(vmax, red);
    float sum = 0.f;
    c = 0;
    for (int j = threadIdx.x; j < S_; j += 256, ++c) {
        float e = expf(vals[c] - vmax);
        vals[c] = e; sum += e;
    }
    sum = block_reduce_sum(sum, red);
    float inv = 1.f / sum;
    c = 0;
    for (int j = threadIdx.x; j < S_; j += 256, ++c)
        sr[j] = vals[c] * inv;
}

// ---------------------------------------------------------------- log-softmax NLL per token
__global__ __launch_bounds__(256) void nll_kernel(
    const float* __restrict__ logits, const int* __restrict__ y,
    float* __restrict__ nll, int bbase) {
    __shared__ float red[256];
    const int s   = blockIdx.x;                 // 0..S-1
    const int tok = bbase + s;
    const float* lr = logits + (long long)s * V_;

    float vmax = -3.4e38f;
    for (int j = threadIdx.x; j < V_; j += 256)
        vmax = fmaxf(vmax, lr[j]);
    vmax = block_reduce_max(vmax, red);
    float sum = 0.f;
    for (int j = threadIdx.x; j < V_; j += 256)
        sum += expf(lr[j] - vmax);
    sum = block_reduce_sum(sum, red);
    if (threadIdx.x == 0) {
        float ly = lr[y[tok]];
        nll[tok] = -(ly - vmax - logf(sum));
    }
}

__global__ __launch_bounds__(256) void mean_kernel(const float* __restrict__ nll,
                                                   float* __restrict__ out) {
    __shared__ float red[256];
    float s = 0.f;
    for (int i = threadIdx.x; i < NT_; i += 256) s += nll[i];
    s = block_reduce_sum(s, red);
    if (threadIdx.x == 0) out[0] = s * (1.f / NT_);
}

// ---------------------------------------------------------------- host side
static void launch_gemm(hipStream_t st,
                        const float* A,  long long lda,  long long aB, long long aH,
                        const float* Bm, long long ldbk, long long ldbn, long long bB, long long bH,
                        float* C,        long long ldc,  long long cB, long long cH,
                        const float* bias, const float* R, long long ldr,
                        int M, int N, int K, int Z, int zdiv, int gelu) {
    GemmParams p;
    p.A = A;  p.lda = lda;  p.aStrB = aB; p.aStrH = aH;
    p.Bm = Bm; p.ldbk = ldbk; p.ldbn = ldbn; p.bStrB = bB; p.bStrH = bH;
    p.C = C;  p.ldc = ldc;  p.cStrB = cB; p.cStrH = cH;
    p.bias = bias; p.R = R; p.ldr = ldr;
    p.M = M; p.N = N; p.K = K; p.zdiv = zdiv; p.gelu = gelu;
    dim3 grid((M + 127) / 128, (N + 63) / 64, Z);
    hipLaunchKernelGGL(gemm_wmma_bf16, grid, dim3(128), 0, st, p);
}

extern "C" void kernel_launch(void* const* d_in, const int* in_sizes, int n_in,
                              void* d_out, int out_size, void* d_ws, size_t ws_size,
                              hipStream_t stream) {
    (void)in_sizes; (void)n_in; (void)out_size; (void)ws_size;
    const int*   x        = (const int*)d_in[0];
    const int*   y        = (const int*)d_in[1];
    const int*   tl       = (const int*)d_in[2];
    const float* word_emb = (const float*)d_in[3];
    const float* pos_emb  = (const float*)d_in[4];
    const float* tok_emb  = (const float*)d_in[5];
    const float* emb_g    = (const float*)d_in[6];
    const float* emb_b    = (const float*)d_in[7];
    const float* Wq       = (const float*)d_in[8];
    const float* bq       = (const float*)d_in[9];
    const float* Wk       = (const float*)d_in[10];
    const float* bk       = (const float*)d_in[11];
    const float* Wv       = (const float*)d_in[12];
    const float* bvv      = (const float*)d_in[13];
    const float* Wo       = (const float*)d_in[14];
    const float* bo       = (const float*)d_in[15];
    const float* ln1g     = (const float*)d_in[16];
    const float* ln1b     = (const float*)d_in[17];
    const float* W1       = (const float*)d_in[18];
    const float* b1       = (const float*)d_in[19];
    const float* W2       = (const float*)d_in[20];
    const float* b2       = (const float*)d_in[21];
    const float* ln2g     = (const float*)d_in[22];
    const float* ln2b     = (const float*)d_in[23];
    const float* clsW     = (const float*)d_in[24];
    const float* clsb     = (const float*)d_in[25];

    // workspace carve-up (fp32)
    float* h      = (float*)d_ws;                                  // NT*H
    float* tmp    = h      + (size_t)NT_ * H_;                     // NT*H
    float* q      = tmp    + (size_t)NT_ * H_;                     // NT*H
    float* k      = q      + (size_t)NT_ * H_;                     // NT*H
    float* v      = k      + (size_t)NT_ * H_;                     // NT*H
    float* scores = v      + (size_t)NT_ * H_;                     // B*NH*S*S
    float* ctx    = scores + (size_t)B_ * NH_ * S_ * S_;           // NT*H
    float* ff     = ctx    + (size_t)NT_ * H_;                     // NT*F
    float* logits = ff     + (size_t)NT_ * F_;                     // S*V
    float* nll    = logits + (size_t)S_ * V_;                      // NT

    // 1) embedding + LN
    hipLaunchKernelGGL(embed_ln_kernel, dim3(NT_), dim3(256), 0, stream,
                       x, word_emb, pos_emb, tok_emb, emb_g, emb_b, h);

    const long long SH = (long long)S_ * H_;
    const long long SS = (long long)S_ * S_;

    for (int l = 0; l < L_; ++l) {
        const float* Wql = Wq + (size_t)l * H_ * H_;
        const float* Wkl = Wk + (size_t)l * H_ * H_;
        const float* Wvl = Wv + (size_t)l * H_ * H_;
        const float* Wol = Wo + (size_t)l * H_ * H_;
        const float* W1l = W1 + (size_t)l * H_ * F_;
        const float* W2l = W2 + (size_t)l * F_ * H_;

        // QKV projections: [NT,H] x [H,H]
        launch_gemm(stream, h, H_, 0, 0, Wql, H_, 1, 0, 0, q, H_, 0, 0,
                    bq + (size_t)l * H_, nullptr, 0, NT_, H_, H_, 1, 1, 0);
        launch_gemm(stream, h, H_, 0, 0, Wkl, H_, 1, 0, 0, k, H_, 0, 0,
                    bk + (size_t)l * H_, nullptr, 0, NT_, H_, H_, 1, 1, 0);
        launch_gemm(stream, h, H_, 0, 0, Wvl, H_, 1, 0, 0, v, H_, 0, 0,
                    bvv + (size_t)l * H_, nullptr, 0, NT_, H_, H_, 1, 1, 0);

        // scores[b,hd,i,j] = q[b,i,hd,:] . k[b,j,hd,:]   (batched z = B*NH)
        launch_gemm(stream,
                    q, H_, SH, DH_,               // A: (i,d), d contiguous
                    k, 1, H_, SH, DH_,            // B: (d,j) = k[j,d] -> ldbk=1, ldbn=H
                    scores, S_, (long long)NH_ * SS, SS,
                    nullptr, nullptr, 0,
                    S_, S_, DH_, B_ * NH_, NH_, 0);

        // masked softmax in place
        hipLaunchKernelGGL(softmax_mask_kernel, dim3(B_ * NH_ * S_), dim3(256), 0, stream,
                           scores, tl);

        // ctx[b,i,hd,:] = probs[b,hd,i,:] @ v[b,:,hd,:]
        launch_gemm(stream,
                    scores, S_, (long long)NH_ * SS, SS,  // A: (i,j)
                    v, H_, 1, SH, DH_,                    // B: (j,d)
                    ctx, H_, SH, DH_,
                    nullptr, nullptr, 0,
                    S_, DH_, S_, B_ * NH_, NH_, 0);

        // attn out + residual, then LN1
        launch_gemm(stream, ctx, H_, 0, 0, Wol, H_, 1, 0, 0, tmp, H_, 0, 0,
                    bo + (size_t)l * H_, h, H_, NT_, H_, H_, 1, 1, 0);
        hipLaunchKernelGGL(ln_kernel, dim3(NT_), dim3(256), 0, stream,
                           tmp, ln1g + (size_t)l * H_, ln1b + (size_t)l * H_, h);

        // FFN
        launch_gemm(stream, h, H_, 0, 0, W1l, F_, 1, 0, 0, ff, F_, 0, 0,
                    b1 + (size_t)l * F_, nullptr, 0, NT_, F_, H_, 1, 1, /*gelu=*/1);
        launch_gemm(stream, ff, F_, 0, 0, W2l, H_, 1, 0, 0, tmp, H_, 0, 0,
                    b2 + (size_t)l * H_, h, H_, NT_, H_, F_, 1, 1, 0);
        hipLaunchKernelGGL(ln_kernel, dim3(NT_), dim3(256), 0, stream,
                           tmp, ln2g + (size_t)l * H_, ln2b + (size_t)l * H_, h);
    }

    // classifier + NLL, chunked per batch (logits scratch = S*V)
    for (int b = 0; b < B_; ++b) {
        launch_gemm(stream, h + (size_t)b * SH, H_, 0, 0,
                    clsW, V_, 1, 0, 0,
                    logits, V_, 0, 0,
                    clsb, nullptr, 0,
                    S_, V_, H_, 1, 1, 0);
        hipLaunchKernelGGL(nll_kernel, dim3(S_), dim3(256), 0, stream,
                           logits, y, nll, b * S_);
    }
    hipLaunchKernelGGL(mean_kernel, dim3(1), dim3(256), 0, stream, nll, (float*)d_out);
}
